// Engine_48378511622701
// MI455X (gfx1250) — matrix operand order repaired and checked by reading, not compiled, not run on hardware
//
#include <hip/hip_runtime.h>

// ---------------------------------------------------------------------------
// DPMPN step on MI455X (gfx1250, wave32, WMMA bf16).
//   prep:  weights -> bf16 col-major; h / node_states -> bf16; gW = gs@Wg
//   gemm:  q,k,mfeat (bf16 out -> halves edge-gather traffic, the roofline
//          bottleneck: ~2.6 GB of gather/scatter vs ~20 us of GEMM work)
//   edges: scores -> segmax -> exp/den -> propagate (attn flow + msg scatter)
//   final: h_new = tanh(ns@Wh + agg@Wa + gW)  -- both GEMMs fused by chaining
//          two v_wmma accumulations into one accumulator per K-step.
// ---------------------------------------------------------------------------

typedef __attribute__((ext_vector_type(16))) __bf16 v16bf;
typedef __attribute__((ext_vector_type(8)))  float  v8f;

static __device__ __forceinline__ __bf16 f2bf(float x) {
    union { float f; unsigned u; } c; c.f = x;
    unsigned r = (c.u + 0x7FFFu + ((c.u >> 16) & 1u)) >> 16;   // RNE
    union { unsigned short s; __bf16 b; } o; o.s = (unsigned short)r;
    return o.b;
}
// unpack two bf16 packed in a uint -> two floats
static __device__ __forceinline__ float2 bfpair(unsigned u) {
    float2 r;
    r.x = __uint_as_float(u << 16);
    r.y = __uint_as_float(u & 0xFFFF0000u);
    return r;
}
// monotonic float<->uint mapping for atomicMax on signed floats
static __device__ __forceinline__ unsigned fkey(float f) {
    unsigned u = __float_as_uint(f);
    return (u & 0x80000000u) ? ~u : (u | 0x80000000u);
}
static __device__ __forceinline__ float funkey(unsigned k) {
    return __uint_as_float((k & 0x80000000u) ? (k ^ 0x80000000u) : ~k);
}

// --------------------------- prep kernels ----------------------------------

// W [128x128] f32 row-major -> Wt bf16 col-major (Wt[col*128+row])
__global__ __launch_bounds__(128)
void prep_w(const float* __restrict__ W, __bf16* __restrict__ Wt) {
    int r = blockIdx.x, c = threadIdx.x;
    Wt[(long)c * 128 + r] = f2bf(W[(long)r * 128 + c]);
}

// gW[b,d] = sum_k gs[b,k] * Wg[k,d]   (tiny: B*D threads, one block)
__global__ __launch_bounds__(256)
void global_proj(const float* __restrict__ gs, const float* __restrict__ Wg,
                 float* __restrict__ gW) {
    int t = threadIdx.x;
    int b = t >> 7, d = t & 127;
    float acc = 0.f;
    #pragma unroll 8
    for (int kk = 0; kk < 128; ++kk) acc += gs[b * 128 + kk] * Wg[kk * 128 + d];
    gW[b * 128 + d] = acc;
}

// hbf = bf16(ns + np), nsbf = bf16(ns); vectorized by 4
__global__ __launch_bounds__(256)
void prep_nodes(const float* __restrict__ ns, const float* __restrict__ np,
                __bf16* __restrict__ hbf, __bf16* __restrict__ nsbf,
                long total, long NDmod) {
    long i = ((long)blockIdx.x * blockDim.x + threadIdx.x) * 4;
    if (i >= total) return;
    float4 a = *(const float4*)(ns + i);
    float4 p = *(const float4*)(np + (i % NDmod));
    hbf[i + 0] = f2bf(a.x + p.x); hbf[i + 1] = f2bf(a.y + p.y);
    hbf[i + 2] = f2bf(a.z + p.z); hbf[i + 3] = f2bf(a.w + p.w);
    nsbf[i + 0] = f2bf(a.x); nsbf[i + 1] = f2bf(a.y);
    nsbf[i + 2] = f2bf(a.z); nsbf[i + 3] = f2bf(a.w);
}

__global__ __launch_bounds__(256)
void zero_f4(float* __restrict__ p, long n4) {
    long i = (long)blockIdx.x * blockDim.x + threadIdx.x;
    if (i < n4) ((float4*)p)[i] = make_float4(0.f, 0.f, 0.f, 0.f);
}

__global__ __launch_bounds__(256)
void init_seg(float* __restrict__ den, unsigned* __restrict__ segmax,
              float* __restrict__ attn_out, long M) {
    long i = (long)blockIdx.x * blockDim.x + threadIdx.x;
    if (i < M) { den[i] = 0.f; segmax[i] = 0x007FFFFFu /* key(-inf) */; attn_out[i] = 0.f; }
}

// --------------------------- WMMA GEMM (bf16 out) --------------------------
// C[M x 128] = A[M x 128](bf16 row-major) @ W(bf16 col-major Wt[n*128+k]).
// Block = 8 waves covering 128 DISTINCT rows x 16 cols (wave w owns rows
// w*16..w*16+15) so A fragments are never loaded redundantly; the 4 KB weight
// tile is the only shared operand (L0-resident). grid = (M/128, D/16).
__global__ __launch_bounds__(256)
void gemm_bf16_qkm(const __bf16* __restrict__ A, const __bf16* __restrict__ Wt,
                   __bf16* __restrict__ C) {
    const int lane = threadIdx.x & 31;
    const int wave = threadIdx.x >> 5;
    const long row0 = (long)blockIdx.x * 128 + wave * 16;
    const int  n0   = blockIdx.y * 16;
    const int  mloc = lane & 15;
    const int  kbA  = (lane < 16) ? 0 : 8;    // ISA 16-bit A 16x32 lane layout
    const int  kbB  = (lane < 16) ? 0 : 8;    // ISA 16-bit B 32x16 lane layout
    const __bf16* arow = A  + (row0 + mloc) * 128;
    const __bf16* bcol = Wt + (long)(n0 + mloc) * 128 + kbB;
    __builtin_prefetch(arow + 128 * 128, 0, 0);  // next block's rows -> global_prefetch_b8
    v8f c = {};
    #pragma unroll
    for (int ks = 0; ks < 128; ks += 32) {
        v16bf a, b;
        *((uint4*)&a)       = *(const uint4*)(arow + ks + kbA);        // K=kbA+0..7
        *(((uint4*)&a) + 1) = *(const uint4*)(arow + ks + kbA + 16);   // K=kbA+16..23
        *((uint4*)&b)       = *(const uint4*)(bcol + ks);              // K=kbB+0..7
        *(((uint4*)&b) + 1) = *(const uint4*)(bcol + ks + 16);         // K=kbB+16..23
        c = __builtin_amdgcn_wmma_f32_16x16x32_bf16(false, a, false, b,
                                                    (short)0, c, false, false);
    }
    const int  col   = n0 + mloc;
    const long rbase = row0 + ((lane < 16) ? 0 : 8);
    #pragma unroll
    for (int v = 0; v < 8; ++v) C[(rbase + v) * 128 + col] = f2bf(c[v]);
}

// --------------------------- edge kernels ----------------------------------

// scores[b,e] = dot(q[b,src], k[b,dst] + rel[etype]) / sqrt(D); segmax per src.
// One wave32 per (b,e): 4 dims/lane, bf16 gathers (8 B/lane/tensor).
__global__ __launch_bounds__(256)
void edge_scores(const __bf16* __restrict__ q, const __bf16* __restrict__ k,
                 const float* __restrict__ etp,
                 const int* __restrict__ src, const int* __restrict__ dst,
                 const int* __restrict__ ety,
                 float* __restrict__ scores, unsigned* __restrict__ segmax,
                 int E, int N) {
    __shared__ float s_et[24 * 128];
    for (int i = threadIdx.x; i < 24 * 128; i += 256) s_et[i] = etp[i];
    __syncthreads();
    const int lane = threadIdx.x & 31;
    const int e = blockIdx.x * 8 + (threadIdx.x >> 5);
    const int b = blockIdx.y;
    if (e >= E) return;
    const int s = src[e], d = dst[e], t = ety[e];
    const uint2 qu = *(const uint2*)(q + ((long)b * N + s) * 128 + lane * 4);
    const uint2 ku = *(const uint2*)(k + ((long)b * N + d) * 128 + lane * 4);
    const float4 rv = *(const float4*)(&s_et[t * 128 + lane * 4]);
    const float2 q01 = bfpair(qu.x), q23 = bfpair(qu.y);
    const float2 k01 = bfpair(ku.x), k23 = bfpair(ku.y);
    float p = q01.x * (k01.x + rv.x) + q01.y * (k01.y + rv.y)
            + q23.x * (k23.x + rv.z) + q23.y * (k23.y + rv.w);
    #pragma unroll
    for (int off = 16; off > 0; off >>= 1) p += __shfl_xor(p, off, 32);
    p *= 0.08838834764831845f;  // 1/sqrt(128)
    if (lane == 0) {
        scores[(long)b * E + e] = p;
        atomicMax(segmax + (long)b * N + s, fkey(p));
    }
}

// ex = exp(score - max[src]); den[src] += ex  (in-place over scores buffer)
__global__ __launch_bounds__(256)
void edge_exp(float* __restrict__ scores, const unsigned* __restrict__ segmax,
              float* __restrict__ den, const int* __restrict__ src,
              long BE, int E, int N) {
    long i = (long)blockIdx.x * blockDim.x + threadIdx.x;
    if (i >= BE) return;
    int b = (int)(i / E), e = (int)(i % E);
    int s = src[e];
    float m = funkey(segmax[(long)b * N + s]);
    float x = __expf(scores[i] - m);
    scores[i] = x;
    atomicAdd(den + (long)b * N + s, x);
}

// trans = ex/den; new_attn[dst] += attn[src]*trans; agg[dst] += trans*(mfeat[src]+rel)
__global__ __launch_bounds__(256)
void edge_propagate(const float* __restrict__ ex, const float* __restrict__ den,
                    const float* __restrict__ attn, const __bf16* __restrict__ mfeat,
                    const float* __restrict__ etp,
                    const int* __restrict__ src, const int* __restrict__ dst,
                    const int* __restrict__ ety,
                    float* __restrict__ new_attn, float* __restrict__ agg,
                    int E, int N) {
    __shared__ float s_et[24 * 128];
    for (int i = threadIdx.x; i < 24 * 128; i += 256) s_et[i] = etp[i];
    __syncthreads();
    const int lane = threadIdx.x & 31;
    const int e = blockIdx.x * 8 + (threadIdx.x >> 5);
    const int b = blockIdx.y;
    if (e >= E) return;
    const int s = src[e], d = dst[e], t = ety[e];
    const float tr = ex[(long)b * E + e] / (den[(long)b * N + s] + 1e-9f);
    if (lane == 0)
        atomicAdd(new_attn + (long)b * N + d, attn[(long)b * N + s] * tr);
    const uint2 mu = *(const uint2*)(mfeat + ((long)b * N + s) * 128 + lane * 4);
    const float4 rv = *(const float4*)(&s_et[t * 128 + lane * 4]);
    const float2 m01 = bfpair(mu.x), m23 = bfpair(mu.y);
    float* ap = agg + ((long)b * N + d) * 128 + lane * 4;
    atomicAdd(ap + 0, tr * (m01.x + rv.x));
    atomicAdd(ap + 1, tr * (m01.y + rv.y));
    atomicAdd(ap + 2, tr * (m23.x + rv.z));
    atomicAdd(ap + 3, tr * (m23.y + rv.w));
}

// --------------------------- fused final GEMMs -----------------------------
// h_new = tanh(ns@Wh + agg@Wa + gW[b]): two WMMA accumulations chained into
// one accumulator per K-step (D = A*B + C), so no hWh intermediate exists.
__global__ __launch_bounds__(256)
void final_update(const float* __restrict__ agg, const __bf16* __restrict__ nsbf,
                  const __bf16* __restrict__ Wht, const __bf16* __restrict__ Wat,
                  const float* __restrict__ gW, float* __restrict__ outH, int N) {
    const int lane = threadIdx.x & 31;
    const int wave = threadIdx.x >> 5;
    const long row0 = (long)blockIdx.x * 128 + wave * 16;
    const int  n0   = blockIdx.y * 16;
    const int  mloc = lane & 15;
    const int  kbA  = (lane < 16) ? 0 : 8;
    const int  kbB  = (lane < 16) ? 0 : 8;
    const float*  grow = agg  + (row0 + mloc) * 128;
    const __bf16* nrow = nsbf + (row0 + mloc) * 128;
    const __bf16* bh   = Wht + (long)(n0 + mloc) * 128 + kbB;
    const __bf16* ba   = Wat + (long)(n0 + mloc) * 128 + kbB;
    v8f c = {};
    #pragma unroll
    for (int ks = 0; ks < 128; ks += 32) {
        // ns @ Wh contribution
        v16bf an, wn;
        *((uint4*)&an)       = *(const uint4*)(nrow + ks + kbA);
        *(((uint4*)&an) + 1) = *(const uint4*)(nrow + ks + kbA + 16);
        *((uint4*)&wn)       = *(const uint4*)(bh + ks);
        *(((uint4*)&wn) + 1) = *(const uint4*)(bh + ks + 16);
        c = __builtin_amdgcn_wmma_f32_16x16x32_bf16(false, an, false, wn,
                                                    (short)0, c, false, false);
        // agg @ Wa contribution (agg is f32; convert on load)
        float4 f0 = *(const float4*)(grow + ks + kbA);
        float4 f1 = *(const float4*)(grow + ks + kbA + 4);
        float4 f2 = *(const float4*)(grow + ks + kbA + 16);
        float4 f3 = *(const float4*)(grow + ks + kbA + 20);
        v16bf aa, wa;
        aa[0]  = f2bf(f0.x); aa[1]  = f2bf(f0.y); aa[2]  = f2bf(f0.z); aa[3]  = f2bf(f0.w);
        aa[4]  = f2bf(f1.x); aa[5]  = f2bf(f1.y); aa[6]  = f2bf(f1.z); aa[7]  = f2bf(f1.w);
        aa[8]  = f2bf(f2.x); aa[9]  = f2bf(f2.y); aa[10] = f2bf(f2.z); aa[11] = f2bf(f2.w);
        aa[12] = f2bf(f3.x); aa[13] = f2bf(f3.y); aa[14] = f2bf(f3.z); aa[15] = f2bf(f3.w);
        *((uint4*)&wa)       = *(const uint4*)(ba + ks);
        *(((uint4*)&wa) + 1) = *(const uint4*)(ba + ks + 16);
        c = __builtin_amdgcn_wmma_f32_16x16x32_bf16(false, aa, false, wa,
                                                    (short)0, c, false, false);
    }
    const int  col   = n0 + mloc;
    const long rbase = row0 + ((lane < 16) ? 0 : 8);
    #pragma unroll
    for (int v = 0; v < 8; ++v) {
        long r = rbase + v;
        int  bb = (int)(r / N);
        outH[r * 128 + col] = tanhf(c[v] + gW[bb * 128 + col]);
    }
}

// --------------------------- launch ----------------------------------------

extern "C" void kernel_launch(void* const* d_in, const int* in_sizes, int n_in,
                              void* d_out, int out_size, void* d_ws, size_t ws_size,
                              hipStream_t stream) {
    const float* attn = (const float*)d_in[0];
    const float* ns   = (const float*)d_in[1];
    const float* np   = (const float*)d_in[2];
    const float* etp  = (const float*)d_in[3];
    const float* gs   = (const float*)d_in[4];
    const float* W[5] = { (const float*)d_in[5], (const float*)d_in[6],
                          (const float*)d_in[7], (const float*)d_in[8],
                          (const float*)d_in[9] };            // Wq Wk Wm Wh Wa
    const float* Wg   = (const float*)d_in[10];
    const int*   src  = (const int*)d_in[11];
    const int*   dst  = (const int*)d_in[12];
    const int*   ety  = (const int*)d_in[13];

    const int  D   = 128;
    const int  N   = in_sizes[2] / D;          // 40000
    const int  B   = in_sizes[0] / N;          // 2
    const int  E   = in_sizes[11];             // 640000
    const long M   = (long)B * N;              // 80000 (multiple of 128)
    const long MND = M * D;

    // workspace carve-up (~149 MB)
    float* ws      = (float*)d_ws;
    float* agg     = ws;                               // MND f32
    float* scores  = agg + MND;                        // B*E f32
    float* den     = scores + (long)B * E;             // M f32
    float* gW      = den + M;                          // B*D f32
    unsigned* segmax = (unsigned*)(gW + (long)B * D);  // M u32
    __bf16* hbf    = (__bf16*)(segmax + M);            // MND bf16
    __bf16* nsbf   = hbf + MND;                        // MND bf16
    __bf16* qbf    = nsbf + MND;                       // MND bf16
    __bf16* kbf    = qbf + MND;                        // MND bf16
    __bf16* mbf    = kbf + MND;                        // MND bf16
    __bf16* wt     = mbf + MND;                        // 5 * D*D bf16

    float* outAttn = (float*)d_out;
    float* outH    = outAttn + M;

    // --- prep ---
    for (int i = 0; i < 5; ++i)
        prep_w<<<128, 128, 0, stream>>>(W[i], wt + (long)i * D * D);
    global_proj<<<1, 256, 0, stream>>>(gs, Wg, gW);
    prep_nodes<<<(int)(MND / 1024), 256, 0, stream>>>(ns, np, hbf, nsbf, MND, (long)N * D);
    zero_f4<<<(int)(MND / 1024), 256, 0, stream>>>(agg, MND / 4);
    init_seg<<<(int)((M + 255) / 256), 256, 0, stream>>>(den, segmax, outAttn, M);

    // --- node GEMMs (WMMA bf16, bf16 outputs for cheap edge gathers) ---
    dim3 gGrid((unsigned)(M / 128), (unsigned)(D / 16));
    gemm_bf16_qkm<<<gGrid, 256, 0, stream>>>(hbf,  wt + 0 * D * D, qbf);
    gemm_bf16_qkm<<<gGrid, 256, 0, stream>>>(hbf,  wt + 1 * D * D, kbf);
    gemm_bf16_qkm<<<gGrid, 256, 0, stream>>>(nsbf, wt + 2 * D * D, mbf);

    // --- edge phase ---
    dim3 eGrid((unsigned)(E / 8), (unsigned)B);
    edge_scores<<<eGrid, 256, 0, stream>>>(qbf, kbf, etp, src, dst, ety, scores, segmax, E, N);
    edge_exp<<<(int)(((long)B * E + 255) / 256), 256, 0, stream>>>(
        scores, segmax, den, src, (long)B * E, E, N);
    edge_propagate<<<eGrid, 256, 0, stream>>>(scores, den, attn, mbf, etp,
                                              src, dst, ety, outAttn, agg, E, N);

    // --- fused final: tanh(ns@Wh + agg@Wa + gW) (8 chained WMMA / wave) ---
    final_update<<<gGrid, 256, 0, stream>>>(agg, nsbf, wt + 3 * D * D,
                                            wt + 4 * D * D, gW, outH, N);
}
